// ResidualSkipRNNForecaster_79370995630585
// MI455X (gfx1250) — compile-verified, matrix operand order checked by reading, not run on hardware
//
#include <hip/hip_runtime.h>
#include <hip/hip_bf16.h>

// ResidualSkipRNNForecaster for MI455X (gfx1250).
// Batch-row-parallel persistent RNN: 8 WGs x 512 threads (16 waves).
// Each WG owns 16 batch rows for all 1024 steps; h0/h1 live in wave registers
// (WMMA f32 D-tile layout), activations staged in LDS as packed bf16 WMMA
// A-operands, weights streamed from L2 as pre-swizzled bf16 B-tiles.
// Round-3 change: x_t is double-buffered through registers — loads for step
// t+1 issue right after the phase-A barrier of step t and complete during the
// GEMM phases, removing the global-load latency from the serial recurrence.

#define B_   128
#define T_   1024
#define DIN  64
#define H_   512
#define EPSC 1e-5f

typedef __attribute__((ext_vector_type(16))) __bf16 v16bf;
typedef __attribute__((ext_vector_type(8)))  float  v8f;

__device__ __forceinline__ unsigned short f2bf(float f) {
    unsigned int u = __float_as_uint(f);
    u += 0x7FFFu + ((u >> 16) & 1u);      // round-to-nearest-even
    return (unsigned short)(u >> 16);
}

// sum across the 16 lanes of each half-wave (wave32 -> two 16-lane groups)
__device__ __forceinline__ float half16_reduce(float v) {
    v += __shfl_xor(v, 8, 16);
    v += __shfl_xor(v, 4, 16);
    v += __shfl_xor(v, 2, 16);
    v += __shfl_xor(v, 1, 16);
    return v;
}

__device__ __forceinline__ v16bf ldsA(const unsigned short* p) {
    return *(const v16bf*)p;
}

// LDS pack index for packed A layout: element (row m, K index kk) of K-tile kt
__device__ __forceinline__ int packIdx(int kt, int m, int kk) {
    int hf = (kk >> 3) & 1;
    int e  = ((kk >> 4) << 3) | (kk & 7);
    return kt * 512 + (hf * 16 + m) * 16 + e;
}

// ---------------------------------------------------------------------------
// Prep: fp32 weights -> bf16, swizzled into the WMMA 32x16 B-operand
// lane/element order (ISA 7.12.2, 16-bit matrices):
//   lane = half*16 + (n & 15);  element e: K = (e<8) ? half*8+e : 16+half*8+(e-8)
// Packed tile (kt,nt) = 512 contiguous bf16; index = ((nt*KT)+kt)*512 + lane*16 + e
// Region offsets in ws: Wi0 @0 (32768), Wh0 @32768, Wi1, Wh1 (262144 each).
// ---------------------------------------------------------------------------
__global__ void pack_weights_kernel(const float* __restrict__ Wi0,
                                    const float* __restrict__ Wh0,
                                    const float* __restrict__ Wi1,
                                    const float* __restrict__ Wh1,
                                    unsigned short* __restrict__ wp) {
    const int mat = blockIdx.y;
    const int KT  = (mat == 0) ? 2 : 16;
    const int nElem = KT * 32 * 512;
    int i = blockIdx.x * blockDim.x + threadIdx.x;
    if (i >= nElem) return;
    const float* W = (mat == 0) ? Wi0 : (mat == 1) ? Wh0 : (mat == 2) ? Wi1 : Wh1;
    size_t off = (mat == 0) ? 0 : (size_t)32768 + (size_t)(mat - 1) * 262144;

    int e    = i & 15;
    int lane = (i >> 4) & 31;
    int tile = i >> 9;
    int kt   = tile % KT;
    int nt   = tile / KT;
    int half = lane >> 4;
    int n    = nt * 16 + (lane & 15);
    int kk   = (e < 8) ? (half * 8 + e) : (16 + half * 8 + (e - 8));
    int k    = kt * 32 + kk;
    wp[off + i] = f2bf(W[(size_t)k * H_ + n]);   // all matrices have N-stride 512
}

// ---------------------------------------------------------------------------
// Persistent RNN kernel.
// ---------------------------------------------------------------------------
__launch_bounds__(512, 1)
__global__ void rnn_persistent_kernel(const float* __restrict__ x,
                                      const float* __restrict__ bi0, const float* __restrict__ bh0,
                                      const float* __restrict__ g0,  const float* __restrict__ be0,
                                      const float* __restrict__ bi1, const float* __restrict__ bh1,
                                      const float* __restrict__ g1,  const float* __restrict__ be1,
                                      const float* __restrict__ Wfc, const float* __restrict__ bfc,
                                      const unsigned short* __restrict__ wp,
                                      float* __restrict__ out) {
    // bf16 A-operand staging, packed per 16x32 K-tile (512 elems = 1KB/tile)
    __shared__ __align__(32) unsigned short h0A[16 * 512];  // 16 KB
    __shared__ __align__(32) unsigned short h1A[16 * 512];  // 16 KB
    __shared__ __align__(32) unsigned short xA [2 * 512];   //  2 KB
    __shared__ float rs0[16], rq0[16], rs1[16], rq1[16];    // LN row sums

    const int tid  = threadIdx.x;
    const int lane = tid & 31;
    const int wid  = tid >> 5;       // 16 waves
    const int hi   = lane >> 4;      // half-wave id
    const int nl   = lane & 15;
    const int rowbase = blockIdx.x * 16;

    const unsigned short* Wi0P = wp;
    const unsigned short* Wh0P = wp + 32768;
    const unsigned short* Wi1P = wp + 32768 + 262144;
    const unsigned short* Wh1P = wp + 32768 + 2 * 262144;

    // Per-tile per-lane constants (each wave owns 2 N-tiles of 16 columns)
    int   ntile[2], cbase[2];
    float bc0[2], g0v[2], be0v[2], bc1[2], g1v[2], be1v[2], wfcv[2];
#pragma unroll
    for (int tt = 0; tt < 2; ++tt) {
        ntile[tt] = wid * 2 + tt;
        int n = ntile[tt] * 16 + nl;
        bc0[tt]  = bi0[n] + bh0[n];
        g0v[tt]  = g0[n];  be0v[tt] = be0[n];
        bc1[tt]  = bi1[n] + bh1[n];
        g1v[tt]  = g1[n];  be1v[tt] = be1[n];
        wfcv[tt] = Wfc[n];
        // where (m, n) of the D-tile lands in the packed A layout (K index = n)
        cbase[tt] = packIdx(n >> 5, 0, n & 31) + hi * 128;   // + r*16 per row
    }

    // Per-wave weight tile bases (element offsets; B tile stride = 512)
    const unsigned short* wi0b0 = Wi0P + (size_t)(ntile[0] * 2)  * 512 + lane * 16;
    const unsigned short* wi0b1 = Wi0P + (size_t)(ntile[1] * 2)  * 512 + lane * 16;
    const unsigned short* wh0b0 = Wh0P + (size_t)(ntile[0] * 16) * 512 + lane * 16;
    const unsigned short* wh0b1 = Wh0P + (size_t)(ntile[1] * 16) * 512 + lane * 16;
    const unsigned short* wi1b0 = Wi1P + (size_t)(ntile[0] * 16) * 512 + lane * 16;
    const unsigned short* wi1b1 = Wi1P + (size_t)(ntile[1] * 16) * 512 + lane * 16;
    const unsigned short* wh1b0 = Wh1P + (size_t)(ntile[0] * 16) * 512 + lane * 16;
    const unsigned short* wh1b1 = Wh1P + (size_t)(ntile[1] * 16) * 512 + lane * 16;

    // x double-buffer: thread handles elements i and i+512 of the 16x64 slab
    const int xm0 = tid >> 6,          xd0 = tid & 63;
    const int xm1 = (tid + 512) >> 6,  xd1 = (tid + 512) & 63;
    const int xp0 = packIdx(xd0 >> 5, xm0, xd0 & 31);
    const int xp1 = packIdx(xd1 >> 5, xm1, xd1 & 31);
    const float* xg0 = x + (size_t)(rowbase + xm0) * T_ * DIN + xd0;  // += DIN per step
    const float* xg1 = x + (size_t)(rowbase + xm1) * T_ * DIN + xd1;
    float xv0 = xg0[0];   // prologue: x at t=0
    float xv1 = xg1[0];

    // zero initial hidden state (registers + LDS staging)
    for (int i = tid; i < 16 * 512; i += 512) { h0A[i] = 0; h1A[i] = 0; }
    v8f h0r[2], h1r[2];
#pragma unroll
    for (int tt = 0; tt < 2; ++tt)
#pragma unroll
        for (int r = 0; r < 8; ++r) { h0r[tt][r] = 0.0f; h1r[tt][r] = 0.0f; }
    __syncthreads();

    for (int t = 0; t < T_; ++t) {
        // ---- phase A: zero LN accumulators, commit prefetched x_t to LDS
        if (tid < 16) { rs0[tid] = 0.f; rq0[tid] = 0.f; rs1[tid] = 0.f; rq1[tid] = 0.f; }
        xA[xp0] = f2bf(xv0);
        xA[xp1] = f2bf(xv1);
        __syncthreads();

        // issue x loads for step t+1 now; they complete during the GEMM phases
        if (t + 1 < T_) {
            xv0 = xg0[(size_t)(t + 1) * DIN];
            xv1 = xg1[(size_t)(t + 1) * DIN];
        }

        // ---- phase B: layer0 GEMM  cand0 = x_t@Wi0 + h0@Wh0 + b
        // kt-outer, both N-tiles share one A load; dual acc chains hide latency.
        v8f acc0, acc1;
#pragma unroll
        for (int r = 0; r < 8; ++r) { acc0[r] = bc0[0]; acc1[r] = bc0[1]; }
#pragma unroll
        for (int kt = 0; kt < 2; ++kt) {
            v16bf a  = ldsA(&xA[kt * 512 + lane * 16]);
            v16bf b0 = *(const v16bf*)(wi0b0 + kt * 512);
            v16bf b1 = *(const v16bf*)(wi0b1 + kt * 512);
            acc0 = __builtin_amdgcn_wmma_f32_16x16x32_bf16(false, a, false, b0,
                                                           (short)0, acc0, false, false);
            acc1 = __builtin_amdgcn_wmma_f32_16x16x32_bf16(false, a, false, b1,
                                                           (short)0, acc1, false, false);
        }
#pragma unroll 4
        for (int kt = 0; kt < 16; ++kt) {
            v16bf a  = ldsA(&h0A[kt * 512 + lane * 16]);
            v16bf b0 = *(const v16bf*)(wh0b0 + kt * 512);
            v16bf b1 = *(const v16bf*)(wh0b1 + kt * 512);
            acc0 = __builtin_amdgcn_wmma_f32_16x16x32_bf16(false, a, false, b0,
                                                           (short)0, acc0, false, false);
            acc1 = __builtin_amdgcn_wmma_f32_16x16x32_bf16(false, a, false, b1,
                                                           (short)0, acc1, false, false);
        }
        v8f v0[2];
#pragma unroll
        for (int r = 0; r < 8; ++r) {
            v0[0][r] = tanhf(acc0[r]) + h0r[0][r];
            v0[1][r] = tanhf(acc1[r]) + h0r[1][r];
        }
        // LN partial sums (per batch row) -> LDS float atomics
#pragma unroll
        for (int r = 0; r < 8; ++r) {
            float s = v0[0][r] + v0[1][r];
            float q = v0[0][r] * v0[0][r] + v0[1][r] * v0[1][r];
            s = half16_reduce(s);
            q = half16_reduce(q);
            if (nl == 0) { atomicAdd(&rs0[r + 8 * hi], s); atomicAdd(&rq0[r + 8 * hi], q); }
        }
        __syncthreads();

        // ---- phase C: LayerNorm0 -> h0 regs + repack bf16 into h0A
#pragma unroll
        for (int tt = 0; tt < 2; ++tt)
#pragma unroll
            for (int r = 0; r < 8; ++r) {
                int m = r + 8 * hi;
                float mu  = rs0[m] * (1.0f / H_);
                float var = rq0[m] * (1.0f / H_) - mu * mu;
                float hn  = (v0[tt][r] - mu) * rsqrtf(var + EPSC) * g0v[tt] + be0v[tt];
                h0r[tt][r] = hn;
                h0A[cbase[tt] + r * 16] = f2bf(hn);
            }
        __syncthreads();

        // ---- phase D: layer1 GEMM  cand1 = h0n@Wi1 + h1@Wh1 + b
#pragma unroll
        for (int r = 0; r < 8; ++r) { acc0[r] = bc1[0]; acc1[r] = bc1[1]; }
#pragma unroll 4
        for (int kt = 0; kt < 16; ++kt) {
            v16bf a  = ldsA(&h0A[kt * 512 + lane * 16]);
            v16bf b0 = *(const v16bf*)(wi1b0 + kt * 512);
            v16bf b1 = *(const v16bf*)(wi1b1 + kt * 512);
            acc0 = __builtin_amdgcn_wmma_f32_16x16x32_bf16(false, a, false, b0,
                                                           (short)0, acc0, false, false);
            acc1 = __builtin_amdgcn_wmma_f32_16x16x32_bf16(false, a, false, b1,
                                                           (short)0, acc1, false, false);
        }
#pragma unroll 4
        for (int kt = 0; kt < 16; ++kt) {
            v16bf a  = ldsA(&h1A[kt * 512 + lane * 16]);
            v16bf b0 = *(const v16bf*)(wh1b0 + kt * 512);
            v16bf b1 = *(const v16bf*)(wh1b1 + kt * 512);
            acc0 = __builtin_amdgcn_wmma_f32_16x16x32_bf16(false, a, false, b0,
                                                           (short)0, acc0, false, false);
            acc1 = __builtin_amdgcn_wmma_f32_16x16x32_bf16(false, a, false, b1,
                                                           (short)0, acc1, false, false);
        }
        v8f v1[2];
#pragma unroll
        for (int r = 0; r < 8; ++r) {
            v1[0][r] = tanhf(acc0[r]) + h1r[0][r];
            v1[1][r] = tanhf(acc1[r]) + h1r[1][r];
        }
#pragma unroll
        for (int r = 0; r < 8; ++r) {
            float s = v1[0][r] + v1[1][r];
            float q = v1[0][r] * v1[0][r] + v1[1][r] * v1[1][r];
            s = half16_reduce(s);
            q = half16_reduce(q);
            if (nl == 0) { atomicAdd(&rs1[r + 8 * hi], s); atomicAdd(&rq1[r + 8 * hi], q); }
        }
        __syncthreads();

        // ---- phase E: LayerNorm1 -> h1 regs + repack bf16 into h1A
#pragma unroll
        for (int tt = 0; tt < 2; ++tt)
#pragma unroll
            for (int r = 0; r < 8; ++r) {
                int m = r + 8 * hi;
                float mu  = rs1[m] * (1.0f / H_);
                float var = rq1[m] * (1.0f / H_) - mu * mu;
                float hn  = (v1[tt][r] - mu) * rsqrtf(var + EPSC) * g1v[tt] + be1v[tt];
                h1r[tt][r] = hn;
                h1A[cbase[tt] + r * 16] = f2bf(hn);
            }
        __syncthreads();
    }

    // ---- output head: out[m] = dot(h1[m,:], Wfc) + bfc  (h1 lives in regs)
    if (tid < 16) rs0[tid] = 0.f;
    __syncthreads();
#pragma unroll
    for (int r = 0; r < 8; ++r) {
        float p = h1r[0][r] * wfcv[0] + h1r[1][r] * wfcv[1];
        p = half16_reduce(p);
        if (nl == 0) atomicAdd(&rs0[r + 8 * hi], p);
    }
    __syncthreads();
    if (tid < 16) out[rowbase + tid] = rs0[tid] + bfc[0];
}

extern "C" void kernel_launch(void* const* d_in, const int* in_sizes, int n_in,
                              void* d_out, int out_size, void* d_ws, size_t ws_size,
                              hipStream_t stream) {
    (void)in_sizes; (void)n_in; (void)out_size; (void)ws_size;
    const float* x   = (const float*)d_in[0];
    const float* Wi0 = (const float*)d_in[1];
    const float* bi0 = (const float*)d_in[2];
    const float* Wh0 = (const float*)d_in[3];
    const float* bh0 = (const float*)d_in[4];
    const float* g0  = (const float*)d_in[5];
    const float* be0 = (const float*)d_in[6];
    const float* Wi1 = (const float*)d_in[7];
    const float* bi1 = (const float*)d_in[8];
    const float* Wh1 = (const float*)d_in[9];
    const float* bh1 = (const float*)d_in[10];
    const float* g1  = (const float*)d_in[11];
    const float* be1 = (const float*)d_in[12];
    const float* Wfc = (const float*)d_in[13];
    const float* bfc = (const float*)d_in[14];
    unsigned short* wp = (unsigned short*)d_ws;   // 1.5625 MB bf16 packed weights
    float* out = (float*)d_out;

    // One-time (per call) weight convert + swizzle: 0.82M elems, trivial cost.
    pack_weights_kernel<<<dim3(1024, 4), 256, 0, stream>>>(Wi0, Wh0, Wi1, Wh1, wp);

    // Persistent scan: 8 workgroups (one per 16 batch rows), 16 waves each.
    rnn_persistent_kernel<<<dim3(B_ / 16), 512, 0, stream>>>(
        x, bi0, bh0, g0, be0, bi1, bh1, g1, be1, Wfc, bfc, wp, out);
}
